// Net_58729382805610
// MI455X (gfx1250) — matrix operand order repaired and checked by reading, not compiled, not run on hardware
//
#include <hip/hip_runtime.h>
#include <hip/hip_bf16.h>

#define NN 100000   // nodes
#define NP 100096   // nodes padded to multiple of 128 (GEMM tile rows)
#define NE 3200000  // edges
#define FIN 128
#define HID 64
#define NC 40
#define NCP 48      // classes padded to 3x16 WMMA tiles

typedef __attribute__((ext_vector_type(16))) __bf16 v16bf;
typedef __attribute__((ext_vector_type(8)))  float  v8f;

__device__ __forceinline__ unsigned short f32_to_bf16_bits(float f) {
    return __builtin_bit_cast(unsigned short, (__bf16)f);   // HW cvt, RNE
}
__device__ __forceinline__ __bf16 bf16_from_bits(unsigned short s) {
    return __builtin_bit_cast(__bf16, s);
}

// ---------------- degree / norm ----------------
__global__ void k_deg_init(float* deg) {
    int i = blockIdx.x * 256 + threadIdx.x;
    if (i < NN) deg[i] = 1.0f;               // self loop weight
}
__global__ void k_deg_scatter(const long long* __restrict__ ei, float* deg) {
    int e = blockIdx.x * 256 + threadIdx.x;
    if (e < NE) atomicAdd(&deg[(int)ei[NE + e]], 1.0f);
}
__global__ void k_deg_rsqrt(float* dinv) {
    int i = blockIdx.x * 256 + threadIdx.x;
    if (i < NN) dinv[i] = rsqrtf(dinv[i]);
}

// ---------------- GEMM 1: h_lin = x @ W1^T  (bf16 WMMA, f32 acc) ----------------
__global__ void __launch_bounds__(256) k_gemm1(const float* __restrict__ x,
                                               const float* __restrict__ W1,
                                               float* __restrict__ hlin) {
    __shared__ unsigned short sW[HID * FIN];     // W1 as bf16 bits, 16 KB
    int tid = threadIdx.x;
    for (int i = tid; i < HID * FIN; i += 256) sW[i] = f32_to_bf16_bits(W1[i]);
    __syncthreads();

    int wave = tid >> 5, lane = tid & 31;
    int ln = lane & 15, g = lane >> 4;
    int row = blockIdx.x * 128 + wave * 16 + ln;       // A row for this lane
    int rowc = row < NN ? row : NN - 1;                // clamp loads only
    const float* xr = x + (long)rowc * FIN;

    v8f zero = {0.f,0.f,0.f,0.f,0.f,0.f,0.f,0.f};
    v8f acc[4]; for (int t = 0; t < 4; ++t) acc[t] = zero;

    #pragma unroll
    for (int kt = 0; kt < 4; ++kt) {
        int kbase = kt * 32;
        v16bf a;
        #pragma unroll
        for (int v = 0; v < 8; ++v) {                  // ISA 16-bit A 16x32 layout
            int k0 = (v < 4) ? (kbase + 2*v + 8*g) : (kbase + 16 + 2*(v-4) + 8*g);
            float2 xv = *(const float2*)(xr + k0);
            a[2*v]   = (__bf16)xv.x;
            a[2*v+1] = (__bf16)xv.y;
        }
        #pragma unroll
        for (int nt = 0; nt < 4; ++nt) {
            v16bf b;
            int n = nt * 16 + ln;                      // ISA 16-bit B 32x16 layout
            #pragma unroll
            for (int v = 0; v < 8; ++v) {
                int k = (v < 4) ? (kbase + 2*v + 8*g) : (kbase + 16 + 2*(v-4) + 8*g);
                b[2*v]   = bf16_from_bits(sW[n * FIN + k]);
                b[2*v+1] = bf16_from_bits(sW[n * FIN + k + 1]);
            }
            acc[nt] = __builtin_amdgcn_wmma_f32_16x16x32_bf16(
                false, a, false, b, (short)0, acc[nt], false, false);
        }
    }
    // unconditional stores into NP-padded buffer (rows >= NN never read)
    int rt = blockIdx.x * 128 + wave * 16;
    #pragma unroll
    for (int nt = 0; nt < 4; ++nt) {
        int ncol = nt * 16 + ln;
        #pragma unroll
        for (int v = 0; v < 8; ++v) {                  // C/D layout: M = v + 8g, N = lane%16
            int m = rt + v + 8*g;
            hlin[(long)m * HID + ncol] = acc[nt][v];
        }
    }
}

// ---------------- layer-1 aggregation ----------------
__global__ void k_selfinit1(const float* __restrict__ dinv, const float* __restrict__ hlin,
                            float* __restrict__ agg) {
    int t = blockIdx.x * 256 + threadIdx.x;            // NN*16 threads, float4 each
    if (t >= NN * 16) return;
    int n = t >> 4, f = (t & 15) * 4;
    float d = dinv[n]; float s = d * d;
    float4 hv = *(const float4*)(hlin + (long)n * HID + f);
    float4 o = { s*hv.x, s*hv.y, s*hv.z, s*hv.w };
    *(float4*)(agg + (long)n * HID + f) = o;
}
__global__ void k_scatter1(const long long* __restrict__ ei, const float* __restrict__ dinv,
                           const float* __restrict__ hlin, float* __restrict__ agg) {
    long long t = (long long)blockIdx.x * 256 + threadIdx.x;  // 16 threads/edge
    long long e = t >> 4;
    if (e >= NE) return;
    int f = (int)(t & 15) * 4;
    int r = (int)ei[e], c = (int)ei[NE + e];
    float nrm = dinv[r] * dinv[c];
    float4 hv = *(const float4*)(hlin + (long)r * HID + f);
    float* dst = agg + (long)c * HID + f;
    atomicAdd(dst + 0, nrm * hv.x);
    atomicAdd(dst + 1, nrm * hv.y);
    atomicAdd(dst + 2, nrm * hv.z);
    atomicAdd(dst + 3, nrm * hv.w);
}
__global__ void k_relu_bf16(const float* __restrict__ agg, const float* __restrict__ b1,
                            unsigned short* __restrict__ hbf) {
    int t = blockIdx.x * 256 + threadIdx.x;
    if (t >= NN * HID) return;
    float v = agg[t] + b1[t & (HID - 1)];
    hbf[t] = f32_to_bf16_bits(v > 0.0f ? v : 0.0f);
}
__global__ void k_pad_hbf(unsigned short* __restrict__ hbf) {
    int t = blockIdx.x * 256 + threadIdx.x;            // zero padded rows
    if (t < (NP - NN) * HID) hbf[NN * HID + t] = 0;
}

// ---------------- GEMM 2: z_lin = h @ W2^T  (40 padded to 48 cols) ----------------
__global__ void __launch_bounds__(256) k_gemm2(const unsigned short* __restrict__ hbf,
                                               const float* __restrict__ W2,
                                               float* __restrict__ zlin) {
    __shared__ unsigned short sW[NCP * HID];           // 6 KB, rows 40..47 zero
    int tid = threadIdx.x;
    for (int i = tid; i < NCP * HID; i += 256) {
        int n = i / HID, k = i % HID;
        sW[i] = f32_to_bf16_bits(n < NC ? W2[n * HID + k] : 0.0f);
    }
    __syncthreads();

    int wave = tid >> 5, lane = tid & 31;
    int ln = lane & 15, g = lane >> 4;
    int row = blockIdx.x * 128 + wave * 16 + ln;       // rows are NP-padded (zeros)
    const unsigned short* hr = hbf + (long)row * HID;

    v8f zero = {0.f,0.f,0.f,0.f,0.f,0.f,0.f,0.f};
    v8f acc[3]; for (int t = 0; t < 3; ++t) acc[t] = zero;

    #pragma unroll
    for (int kt = 0; kt < 2; ++kt) {
        int kbase = kt * 32;
        v16bf a;
        #pragma unroll
        for (int v = 0; v < 8; ++v) {
            int k0 = (v < 4) ? (kbase + 2*v + 8*g) : (kbase + 16 + 2*(v-4) + 8*g);
            unsigned int w = *(const unsigned int*)(hr + k0);
            a[2*v]   = bf16_from_bits((unsigned short)(w & 0xFFFFu));
            a[2*v+1] = bf16_from_bits((unsigned short)(w >> 16));
        }
        #pragma unroll
        for (int nt = 0; nt < 3; ++nt) {
            v16bf b;
            int n = nt * 16 + ln;
            #pragma unroll
            for (int v = 0; v < 8; ++v) {
                int k = (v < 4) ? (kbase + 2*v + 8*g) : (kbase + 16 + 2*(v-4) + 8*g);
                b[2*v]   = bf16_from_bits(sW[n * HID + k]);
                b[2*v+1] = bf16_from_bits(sW[n * HID + k + 1]);
            }
            acc[nt] = __builtin_amdgcn_wmma_f32_16x16x32_bf16(
                false, a, false, b, (short)0, acc[nt], false, false);
        }
    }
    int rt = blockIdx.x * 128 + wave * 16;
    #pragma unroll
    for (int nt = 0; nt < 3; ++nt) {
        int ncol = nt * 16 + ln;
        #pragma unroll
        for (int v = 0; v < 8; ++v) {
            int m = rt + v + 8*g;
            zlin[(long)m * NCP + ncol] = acc[nt][v];   // NP-padded, unguarded
        }
    }
}

// ---------------- layer-2 aggregation ----------------
__global__ void k_selfinit2(const float* __restrict__ dinv, const float* __restrict__ zlin,
                            float* __restrict__ zagg) {
    int t = blockIdx.x * 256 + threadIdx.x;
    if (t >= NN * NC) return;
    int n = t / NC, c = t - n * NC;
    float d = dinv[n];
    zagg[t] = d * d * zlin[(long)n * NCP + c];
}
__global__ void k_scatter2(const long long* __restrict__ ei, const float* __restrict__ dinv,
                           const float* __restrict__ zlin, float* __restrict__ zagg) {
    long long t = (long long)blockIdx.x * 256 + threadIdx.x;  // 8 threads/edge, 5 cols each
    long long e = t >> 3;
    if (e >= NE) return;
    int f = (int)(t & 7) * 5;
    int r = (int)ei[e], c = (int)ei[NE + e];
    float nrm = dinv[r] * dinv[c];
    const float* src = zlin + (long)r * NCP + f;
    float* dst = zagg + (long)c * NC + f;
    #pragma unroll
    for (int j = 0; j < 5; ++j) atomicAdd(dst + j, nrm * src[j]);
}

// ---------------- softmax / log-softmax / argmax ----------------
__global__ void k_softmax(const float* __restrict__ zagg, const float* __restrict__ b2,
                          float* __restrict__ out, int* __restrict__ pred) {
    int n = blockIdx.x * 256 + threadIdx.x;
    if (n >= NN) return;
    float z[NC];
    float mx = -1e30f; int am = 0;
    #pragma unroll
    for (int c = 0; c < NC; ++c) {
        z[c] = zagg[(long)n * NC + c] + b2[c];
        if (z[c] > mx) { mx = z[c]; am = c; }
    }
    float s = 0.0f;
    #pragma unroll
    for (int c = 0; c < NC; ++c) s += __expf(z[c] - mx);
    float ls = __logf(s), inv = 1.0f / s;
    #pragma unroll
    for (int c = 0; c < NC; ++c) {
        float e = z[c] - mx;
        out[(long)n * NC + c] = e - ls;                          // logp
        out[(long)NN * NC + (long)n * NC + c] = __expf(e) * inv; // p
    }
    pred[n] = am;
}

// ---------------- homophily ratio ----------------
__global__ void k_zero_cnt(unsigned int* cnt) { if (threadIdx.x == 0) *cnt = 0u; }
__global__ void k_ratio(const long long* __restrict__ ei, const long long* __restrict__ y,
                        const int* __restrict__ pred, unsigned int* cnt) {
    int e = blockIdx.x * 256 + threadIdx.x;
    if (e >= NE) return;
    int r = (int)ei[e], c = (int)ei[NE + e];
    bool lab = (y[r] == y[c]);
    bool agree = ((pred[r] == pred[c]) == lab);
    if (agree) atomicAdd(cnt, 1u);
}
__global__ void k_finalize(const unsigned int* cnt, float* out) {
    if (threadIdx.x == 0) out[(long)2 * NN * NC] = (float)(*cnt) / (float)NE;
}

static inline size_t alignup(size_t x) { return (x + 255) & ~(size_t)255; }

extern "C" void kernel_launch(void* const* d_in, const int* in_sizes, int n_in,
                              void* d_out, int out_size, void* d_ws, size_t ws_size,
                              hipStream_t stream) {
    const float*     x  = (const float*)d_in[0];
    const long long* ei = (const long long*)d_in[1];
    const long long* y  = (const long long*)d_in[2];
    const float*     W1 = (const float*)d_in[3];
    const float*     b1 = (const float*)d_in[4];
    const float*     W2 = (const float*)d_in[5];
    const float*     b2 = (const float*)d_in[6];
    float* out = (float*)d_out;

    char* ws = (char*)d_ws;
    size_t off = 0;
    float* dinv = (float*)(ws + off); off = alignup(off + (size_t)NN * 4);
    float* bufA = (float*)(ws + off); off = alignup(off + (size_t)NP * HID * 4); // hlin -> zagg
    float* bufB = (float*)(ws + off); off = alignup(off + (size_t)NP * HID * 4); // agg1 -> zlin
    unsigned short* hbf = (unsigned short*)(ws + off); off = alignup(off + (size_t)NP * HID * 2);
    int* pred = (int*)(ws + off); off = alignup(off + (size_t)NN * 4);
    unsigned int* cnt = (unsigned int*)(ws + off);

    float* hlin = bufA;
    float* agg1 = bufB;
    float* zlin = bufB;   // reuse after agg1 dead (NP*48 <= NP*64)
    float* zagg = bufA;   // reuse after hlin dead (NN*40 <= NP*64)

    const int TPB = 256;
    // degree norm
    k_deg_init<<<(NN + TPB - 1) / TPB, TPB, 0, stream>>>(dinv);
    k_deg_scatter<<<(NE + TPB - 1) / TPB, TPB, 0, stream>>>(ei, dinv);
    k_deg_rsqrt<<<(NN + TPB - 1) / TPB, TPB, 0, stream>>>(dinv);
    // layer 1
    k_gemm1<<<NP / 128, TPB, 0, stream>>>(x, W1, hlin);
    k_selfinit1<<<(NN * 16 + TPB - 1) / TPB, TPB, 0, stream>>>(dinv, hlin, agg1);
    k_scatter1<<<(int)(((long long)NE * 16 + TPB - 1) / TPB), TPB, 0, stream>>>(ei, dinv, hlin, agg1);
    k_relu_bf16<<<(NN * HID + TPB - 1) / TPB, TPB, 0, stream>>>(agg1, b1, hbf);
    k_pad_hbf<<<((NP - NN) * HID + TPB - 1) / TPB, TPB, 0, stream>>>(hbf);
    // layer 2
    k_gemm2<<<NP / 128, TPB, 0, stream>>>(hbf, W2, zlin);
    k_selfinit2<<<(NN * NC + TPB - 1) / TPB, TPB, 0, stream>>>(dinv, zlin, zagg);
    k_scatter2<<<(int)(((long long)NE * 8 + TPB - 1) / TPB), TPB, 0, stream>>>(ei, dinv, zlin, zagg);
    // outputs
    k_softmax<<<(NN + TPB - 1) / TPB, TPB, 0, stream>>>(zagg, b2, out, pred);
    k_zero_cnt<<<1, 1, 0, stream>>>(cnt);
    k_ratio<<<(NE + TPB - 1) / TPB, TPB, 0, stream>>>(ei, y, pred, cnt);
    k_finalize<<<1, 1, 0, stream>>>(cnt, out);
}
